// PortfolioConstructor_27754078667250
// MI455X (gfx1250) — compile-verified, matrix operand order checked by reading, not run on hardware
//
#include <hip/hip_runtime.h>

// ---------------------------------------------------------------------------
// Types
// ---------------------------------------------------------------------------
typedef __bf16 bf16_t;
typedef __attribute__((ext_vector_type(16))) __bf16 v16bf;
typedef __attribute__((ext_vector_type(8)))  __bf16 bf16x8;
typedef __attribute__((ext_vector_type(8)))  float  v8f;

#define DEV __device__ __forceinline__

DEV v8f zero8() {
  v8f z = {0.f, 0.f, 0.f, 0.f, 0.f, 0.f, 0.f, 0.f};
  return z;
}

// D = A(16x32) * B(32x16) + C, bf16 inputs, f32 accumulate
DEV v8f wmma_bf16(v16bf a, v16bf b, v8f c) {
  return __builtin_amdgcn_wmma_f32_16x16x32_bf16(
      /*neg_a=*/false, a, /*neg_b=*/false, b,
      /*c_mod=*/(short)0, c, /*reuse_a=*/false, /*reuse_b=*/false);
}

// Load a 16x32 (or 32x16-as-rows) bf16 fragment.  Caller passes the per-lane
// base pointer p = base + (row0 + lane%16)*ld + k0 + 8*(lane/16).
// Per CDNA5 16-bit A/B layout: halfwords [0..7] = K..K+7, [8..15] = K+16..K+23.
DEV v16bf frag_rowmajor(const bf16_t* p) {
  bf16x8 lo = *(const bf16x8*)p;
  bf16x8 hi = *(const bf16x8*)(p + 16);
  v16bf f;
#pragma unroll
  for (int i = 0; i < 8; ++i) { f[i] = lo[i]; f[i + 8] = hi[i]; }
  return f;
}

DEV float wave_reduce_sum(float v) {
#pragma unroll
  for (int m = 1; m < 32; m <<= 1) v += __shfl_xor(v, m, 32);
  return v;
}
DEV float half_reduce_max(float v) {  // reduce within each 16-lane half
#pragma unroll
  for (int m = 1; m < 16; m <<= 1) v = fmaxf(v, __shfl_xor(v, m, 32));
  return v;
}
DEV float half_reduce_sum(float v) {
#pragma unroll
  for (int m = 1; m < 16; m <<= 1) v += __shfl_xor(v, m, 32);
  return v;
}

// ---------------------------------------------------------------------------
// f32 -> bf16 conversion
// ---------------------------------------------------------------------------
__global__ void f32_to_bf16_kernel(const float* __restrict__ in,
                                   bf16_t* __restrict__ out, int n) {
  int i = blockIdx.x * blockDim.x + threadIdx.x;
  if (i < n) out[i] = (bf16_t)in[i];
}

// ---------------------------------------------------------------------------
// Generic GEMM: C[M,N] = scale * (A[M,K] @ W[N,K]^T) + bias, optional ReLU.
// A, W bf16 row-major; one wave computes a 16(M) x 64(N) tile via WMMA.
// ---------------------------------------------------------------------------
__global__ void gemm_bf16_wmma(const bf16_t* __restrict__ A,
                               const bf16_t* __restrict__ W,
                               const float* __restrict__ bias,
                               float* __restrict__ Cf, bf16_t* __restrict__ Cb,
                               int M, int K, int N, float scale, int relu) {
  const int lane = threadIdx.x & 31;
  const int wave = blockIdx.x * (blockDim.x >> 5) + (threadIdx.x >> 5);
  const int tiles_n = N >> 6;
  const int mt = wave / tiles_n;
  if (mt * 16 >= M) return;          // wave-uniform: EXEC stays all-ones
  const int nt = wave - mt * tiles_n;
  const int m0 = mt << 4, n0 = nt << 6;
  const int rsel = lane & 15;
  const int ksel = (lane >> 4) << 3;

  v8f acc[4] = {zero8(), zero8(), zero8(), zero8()};
  const bf16_t* ap = A + (size_t)(m0 + rsel) * K + ksel;
  const bf16_t* wp = W + (size_t)(n0 + rsel) * K + ksel;
  for (int k = 0; k < K; k += 32) {
    v16bf a  = frag_rowmajor(ap + k);
    v16bf b0 = frag_rowmajor(wp + k);
    v16bf b1 = frag_rowmajor(wp + (size_t)16 * K + k);
    v16bf b2 = frag_rowmajor(wp + (size_t)32 * K + k);
    v16bf b3 = frag_rowmajor(wp + (size_t)48 * K + k);
    acc[0] = wmma_bf16(a, b0, acc[0]);
    acc[1] = wmma_bf16(a, b1, acc[1]);
    acc[2] = wmma_bf16(a, b2, acc[2]);
    acc[3] = wmma_bf16(a, b3, acc[3]);
  }
  // C/D layout: lane l, VGPR r -> row = r + 8*(l>=16), col = l%16
  const int rowoff = (lane >> 4) << 3;
#pragma unroll
  for (int j = 0; j < 4; ++j) {
    const int col = n0 + 16 * j + rsel;
    const float bsv = bias ? bias[col] : 0.f;
#pragma unroll
    for (int r = 0; r < 8; ++r) {
      const int row = m0 + rowoff + r;
      float v = acc[j][r] * scale + bsv;
      if (relu) v = fmaxf(v, 0.f);
      size_t o = (size_t)row * N + col;
      if (Cf) Cf[o] = v;
      if (Cb) Cb[o] = (bf16_t)v;
    }
  }
}

// ---------------------------------------------------------------------------
// Fused per-(asset, head) attention: S=64, hd=32.
// qkv: [64, 1024, 768] bf16 (q | k | v each 256 wide, head h at h*32).
// o:   [64, 1024, 256] bf16.
// Block = 128 threads (4 waves); wave w handles query rows [16w, 16w+16).
// ---------------------------------------------------------------------------
#define SQK 40   // Q/K LDS row stride (halfwords): 80B, 16B-aligned, no bank conflicts
#define SPV 72   // Vt/P LDS row stride (halfwords): 144B, 16B-aligned, no bank conflicts

__global__ void attn64_kernel(const bf16_t* __restrict__ qkv,
                              bf16_t* __restrict__ o) {
  const int n = blockIdx.x;   // asset
  const int h = blockIdx.y;   // head
  __shared__ __attribute__((aligned(16))) bf16_t Qs[64 * SQK];
  __shared__ __attribute__((aligned(16))) bf16_t Ks[64 * SQK];
  __shared__ __attribute__((aligned(16))) bf16_t Vt[32 * SPV];     // [d][t]
  __shared__ __attribute__((aligned(16))) bf16_t Pw[4][16 * SPV];  // per-wave P

  const int tid = threadIdx.x;                 // 0..127
  const size_t tokStride = (size_t)1024 * 768; // seq stride in qkv
  const bf16_t* base = qkv + (size_t)n * 768 + h * 32;
  {
    const int s = tid >> 1, half = tid & 1;    // 64 rows x 2 halves
    const bf16_t* g = base + (size_t)s * tokStride + half * 16;
    *(bf16x8*)&Qs[s * SQK + half * 16]     = *(const bf16x8*)(g);
    *(bf16x8*)&Qs[s * SQK + half * 16 + 8] = *(const bf16x8*)(g + 8);
    *(bf16x8*)&Ks[s * SQK + half * 16]     = *(const bf16x8*)(g + 256);
    *(bf16x8*)&Ks[s * SQK + half * 16 + 8] = *(const bf16x8*)(g + 264);
    bf16x8 va = *(const bf16x8*)(g + 512);
    bf16x8 vb = *(const bf16x8*)(g + 520);
#pragma unroll
    for (int i = 0; i < 8; ++i) {              // transpose V into Vt[d][t]
      Vt[(half * 16 + i) * SPV + s]     = va[i];
      Vt[(half * 16 + 8 + i) * SPV + s] = vb[i];
    }
  }
  __syncthreads();

  const int wv = tid >> 5, lane = tid & 31;
  const int s0 = wv << 4;
  const int rsel = lane & 15;
  const int ksel = (lane >> 4) << 3;
  const int rowoff = (lane >> 4) << 3;

  // scores[16,64] = Q_tile(16x32) @ K^T(32x64)
  v8f c[4] = {zero8(), zero8(), zero8(), zero8()};
  {
    v16bf a = frag_rowmajor(&Qs[(s0 + rsel) * SQK + ksel]);
#pragma unroll
    for (int j = 0; j < 4; ++j) {
      v16bf b = frag_rowmajor(&Ks[(16 * j + rsel) * SQK + ksel]);
      c[j] = wmma_bf16(a, b, c[j]);
    }
  }

  // softmax over 64 columns per row; row r lives in one 16-lane half
  const float sc = 0.17677669529663687f;  // 1/sqrt(32)
#pragma unroll
  for (int r = 0; r < 8; ++r) {
    float m = -3.0e38f;
#pragma unroll
    for (int j = 0; j < 4; ++j) m = fmaxf(m, c[j][r] * sc);
    m = half_reduce_max(m);
    float s = 0.f;
#pragma unroll
    for (int j = 0; j < 4; ++j) {
      float e = __expf(c[j][r] * sc - m);
      c[j][r] = e;
      s += e;
    }
    s = half_reduce_sum(s);
    const float inv = 1.0f / s;
    const int lr = rowoff + r;  // local row within the 16-row tile
#pragma unroll
    for (int j = 0; j < 4; ++j)
      Pw[wv][lr * SPV + 16 * j + rsel] = (bf16_t)(c[j][r] * inv);
  }

  // O[16,32] = P(16x64) @ V(64x32); wave-private LDS, DS ops are in-order
  v8f oa[2] = {zero8(), zero8()};
#pragma unroll
  for (int ks = 0; ks < 2; ++ks) {
    v16bf p = frag_rowmajor(&Pw[wv][rsel * SPV + 32 * ks + ksel]);
#pragma unroll
    for (int jn = 0; jn < 2; ++jn) {
      v16bf vv = frag_rowmajor(&Vt[(16 * jn + rsel) * SPV + 32 * ks + ksel]);
      oa[jn] = wmma_bf16(p, vv, oa[jn]);
    }
  }
#pragma unroll
  for (int jn = 0; jn < 2; ++jn) {
#pragma unroll
    for (int r = 0; r < 8; ++r) {
      const int sr = s0 + rowoff + r;
      const int d = 16 * jn + rsel;
      o[((size_t)sr * 1024 + n) * 256 + h * 32 + d] = (bf16_t)oa[jn][r];
    }
  }
}

// ---------------------------------------------------------------------------
// LayerNorm over F=64 features (pre-embedding), one wave per token
// ---------------------------------------------------------------------------
__global__ void ln_feat_kernel(const float* __restrict__ x,
                               const float* __restrict__ g,
                               const float* __restrict__ b,
                               bf16_t* __restrict__ out) {
  const int tok = blockIdx.x * (blockDim.x >> 5) + (threadIdx.x >> 5);
  const int lane = threadIdx.x & 31;
  const float* xp = x + (size_t)tok * 64;
  float a = xp[lane], c2 = xp[lane + 32];
  float mean = wave_reduce_sum(a + c2) * (1.f / 64.f);
  float da = a - mean, db = c2 - mean;
  float var = wave_reduce_sum(da * da + db * db) * (1.f / 64.f);
  float inv = rsqrtf(var + 1e-5f);
  bf16_t* op = out + (size_t)tok * 64;
  op[lane]      = (bf16_t)(da * inv * g[lane] + b[lane]);
  op[lane + 32] = (bf16_t)(db * inv * g[lane + 32] + b[lane + 32]);
}

// ---------------------------------------------------------------------------
// h = LN(h + res) over D=256; one wave per token; writes f32 + bf16 copies
// ---------------------------------------------------------------------------
__global__ void ln_res_kernel(const float* __restrict__ hin,
                              const float* __restrict__ res,
                              const float* __restrict__ g,
                              const float* __restrict__ b,
                              float* __restrict__ hout,
                              bf16_t* __restrict__ hbf) {
  const int tok = blockIdx.x * (blockDim.x >> 5) + (threadIdx.x >> 5);
  const int lane = threadIdx.x & 31;
  const size_t base = (size_t)tok * 256;
  float v[8];
  float s = 0.f;
#pragma unroll
  for (int i = 0; i < 8; ++i) {
    const int idx = lane + 32 * i;
    v[i] = hin[base + idx] + res[base + idx];
    s += v[i];
  }
  const float mean = wave_reduce_sum(s) * (1.f / 256.f);
  float var = 0.f;
#pragma unroll
  for (int i = 0; i < 8; ++i) { v[i] -= mean; var += v[i] * v[i]; }
  var = wave_reduce_sum(var) * (1.f / 256.f);
  const float inv = rsqrtf(var + 1e-5f);
#pragma unroll
  for (int i = 0; i < 8; ++i) {
    const int idx = lane + 32 * i;
    const float y = v[i] * inv * g[idx] + b[idx];
    hout[base + idx] = y;
    hbf[base + idx] = (bf16_t)y;
  }
}

// ---------------------------------------------------------------------------
// r[n,d] = mean_s h[s,n,d]  (h: [64, 262144] with seq stride 262144)
// ---------------------------------------------------------------------------
__global__ void mean_s_kernel(const float* __restrict__ h,
                              bf16_t* __restrict__ r) {
  const int idx = blockIdx.x * blockDim.x + threadIdx.x;  // < 262144
  float s = 0.f;
#pragma unroll 8
  for (int t = 0; t < 64; ++t) s += h[(size_t)t * 262144 + idx];
  r[idx] = (bf16_t)(s * (1.f / 64.f));
}

// ---------------------------------------------------------------------------
// Row softmax over 1024 columns -> bf16 P
// ---------------------------------------------------------------------------
__global__ void softmax1024_kernel(const float* __restrict__ S,
                                   bf16_t* __restrict__ P) {
  __shared__ float red[256];
  const int row = blockIdx.x, tid = threadIdx.x;
  const float* sr = S + (size_t)row * 1024;
  float v[4];
  float m = -3.0e38f;
#pragma unroll
  for (int i = 0; i < 4; ++i) { v[i] = sr[tid + 256 * i]; m = fmaxf(m, v[i]); }
  red[tid] = m; __syncthreads();
  for (int o2 = 128; o2 > 0; o2 >>= 1) {
    if (tid < o2) red[tid] = fmaxf(red[tid], red[tid + o2]);
    __syncthreads();
  }
  m = red[0]; __syncthreads();
  float s = 0.f;
#pragma unroll
  for (int i = 0; i < 4; ++i) { v[i] = __expf(v[i] - m); s += v[i]; }
  red[tid] = s; __syncthreads();
  for (int o2 = 128; o2 > 0; o2 >>= 1) {
    if (tid < o2) red[tid] += red[tid + o2];
    __syncthreads();
  }
  const float inv = 1.f / red[0];
  bf16_t* pr = P + (size_t)row * 1024;
#pragma unroll
  for (int i = 0; i < 4; ++i) pr[tid + 256 * i] = (bf16_t)(v[i] * inv);
}

// vT[64,1024] = v[1024,64]^T
__global__ void transpose_v_kernel(const bf16_t* __restrict__ v,
                                   bf16_t* __restrict__ vT) {
  const int t = blockIdx.x * blockDim.x + threadIdx.x;  // < 65536
  const int n = t >> 6, d = t & 63;
  vT[(size_t)d * 1024 + n] = v[t];
}

// scores[n] = av[n,:] . Ws + bs
__global__ void score_kernel(const float* __restrict__ av,
                             const float* __restrict__ Ws,
                             const float* __restrict__ bs,
                             float* __restrict__ sc) {
  const int n = blockIdx.x * blockDim.x + threadIdx.x;
  if (n >= 1024) return;
  float s = bs[0];
#pragma unroll 8
  for (int j = 0; j < 64; ++j) s += av[(size_t)n * 64 + j] * Ws[j];
  sc[n] = s;
}

// ---------------------------------------------------------------------------
// Top-10 selection + masked softmax allocation.
// out[0..9]  = sorted top-10 indices (as floats)
// out[10..1033] = softmax(scores - 1e9*(1 - selected))
// ---------------------------------------------------------------------------
__global__ void topk_alloc_kernel(const float* __restrict__ scores,
                                  float* __restrict__ out) {
  const int tid = threadIdx.x;  // 1024
  __shared__ float rv[1024];
  __shared__ int ri[1024];
  __shared__ float cur[1024];
  __shared__ int sel[1024];
  __shared__ int chosen[10];
  const float sc = scores[tid];
  cur[tid] = sc;
  sel[tid] = 0;
  __syncthreads();
  for (int k = 0; k < 10; ++k) {
    rv[tid] = cur[tid]; ri[tid] = tid; __syncthreads();
    for (int o2 = 512; o2 > 0; o2 >>= 1) {
      if (tid < o2) {
        const float a = rv[tid], bb = rv[tid + o2];
        const int ia = ri[tid], ib = ri[tid + o2];
        if (bb > a || (bb == a && ib < ia)) { rv[tid] = bb; ri[tid] = ib; }
      }
      __syncthreads();
    }
    if (tid == 0) {
      const int w = ri[0];
      chosen[k] = w;
      sel[w] = 1;
      cur[w] = -3.0e38f;
    }
    __syncthreads();
  }
  if (tid == 0) {
    for (int i = 1; i < 10; ++i) {  // insertion sort ascending
      const int key = chosen[i];
      int j = i - 1;
      while (j >= 0 && chosen[j] > key) { chosen[j + 1] = chosen[j]; --j; }
      chosen[j + 1] = key;
    }
    for (int i = 0; i < 10; ++i) out[i] = (float)chosen[i];
  }
  __syncthreads();
  const float z = sc - (sel[tid] ? 0.f : 1e9f);
  rv[tid] = z; __syncthreads();
  for (int o2 = 512; o2 > 0; o2 >>= 1) {
    if (tid < o2) rv[tid] = fmaxf(rv[tid], rv[tid + o2]);
    __syncthreads();
  }
  const float m = rv[0]; __syncthreads();
  const float e = __expf(z - m);
  rv[tid] = e; __syncthreads();
  for (int o2 = 512; o2 > 0; o2 >>= 1) {
    if (tid < o2) rv[tid] += rv[tid + o2];
    __syncthreads();
  }
  out[10 + tid] = e / rv[0];
}

// ---------------------------------------------------------------------------
// Host orchestration
// ---------------------------------------------------------------------------
extern "C" void kernel_launch(void* const* d_in, const int* in_sizes, int n_in,
                              void* d_out, int out_size, void* d_ws, size_t ws_size,
                              hipStream_t stream) {
  (void)in_sizes; (void)n_in; (void)out_size; (void)ws_size;
  const float* x    = (const float*)d_in[0];
  const float* g0   = (const float*)d_in[1];
  const float* b0   = (const float*)d_in[2];
  const float* Wc   = (const float*)d_in[3];
  const float* bc   = (const float*)d_in[4];
  const float* Wqkv = (const float*)d_in[5];
  const float* bqkv = (const float*)d_in[6];
  const float* Wo   = (const float*)d_in[7];
  const float* bo   = (const float*)d_in[8];
  const float* ln1g = (const float*)d_in[9];
  const float* ln1b = (const float*)d_in[10];
  const float* W1   = (const float*)d_in[11];
  const float* b1   = (const float*)d_in[12];
  const float* W2   = (const float*)d_in[13];
  const float* b2   = (const float*)d_in[14];
  const float* ln2g = (const float*)d_in[15];
  const float* ln2b = (const float*)d_in[16];
  const float* WQ   = (const float*)d_in[17];
  const float* bQ   = (const float*)d_in[18];
  const float* WK   = (const float*)d_in[19];
  const float* bK   = (const float*)d_in[20];
  const float* WV   = (const float*)d_in[21];
  const float* bV   = (const float*)d_in[22];
  const float* Wsw  = (const float*)d_in[23];
  const float* bsb  = (const float*)d_in[24];

  const int TOK = 64 * 1024;  // 65536 tokens

  // ---- workspace layout (offsets 256B aligned) ----
  char* wsb = (char*)d_ws;
  size_t off = 0;
  auto AL = [&](size_t bytes) -> char* {
    char* p = wsb + off;
    off += (bytes + 255) & ~(size_t)255;
    return p;
  };
  bf16_t* hn_bf  = (bf16_t*)AL((size_t)TOK * 64 * 2);    // normalized input
  float*  h_f32  = (float*) AL((size_t)TOK * 256 * 4);   // residual stream
  bf16_t* h_bf   = (bf16_t*)AL((size_t)TOK * 256 * 2);
  char*   big    = AL((size_t)TOK * 768 * 2);            // qkv bf16 / tmp f32 alias
  bf16_t* qkv_bf = (bf16_t*)big;
  float*  tmp_f32= (float*)big;
  bf16_t* o_bf   = (bf16_t*)AL((size_t)TOK * 256 * 2);   // attn out / ff1 alias
  bf16_t* ff1_bf = o_bf;
  bf16_t* wc_bf   = (bf16_t*)AL((size_t)256 * 64 * 2);
  bf16_t* wqkv_bf = (bf16_t*)AL((size_t)4 * 768 * 256 * 2);
  bf16_t* wo_bf   = (bf16_t*)AL((size_t)4 * 256 * 256 * 2);
  bf16_t* w1_bf   = (bf16_t*)AL((size_t)4 * 128 * 256 * 2);
  bf16_t* w2_bf   = (bf16_t*)AL((size_t)4 * 256 * 128 * 2);
  bf16_t* wq_bf   = (bf16_t*)AL((size_t)64 * 256 * 2);
  bf16_t* wk_bf   = (bf16_t*)AL((size_t)64 * 256 * 2);
  bf16_t* wv_bf   = (bf16_t*)AL((size_t)64 * 256 * 2);
  bf16_t* r_bf    = (bf16_t*)AL((size_t)1024 * 256 * 2);
  bf16_t* cq_bf   = (bf16_t*)AL((size_t)1024 * 64 * 2);
  bf16_t* ck_bf   = (bf16_t*)AL((size_t)1024 * 64 * 2);
  bf16_t* cv_bf   = (bf16_t*)AL((size_t)1024 * 64 * 2);
  bf16_t* cvT_bf  = (bf16_t*)AL((size_t)64 * 1024 * 2);
  float*  attnS   = (float*) AL((size_t)1024 * 1024 * 4);
  bf16_t* attnP   = (bf16_t*)AL((size_t)1024 * 1024 * 2);
  float*  av      = (float*) AL((size_t)1024 * 64 * 4);
  float*  scores  = (float*) AL((size_t)1024 * 4);

  auto conv = [&](const float* src, bf16_t* dst, int n) {
    f32_to_bf16_kernel<<<(n + 255) / 256, 256, 0, stream>>>(src, dst, n);
  };
  auto gemm = [&](const bf16_t* A, const bf16_t* Wm, const float* bias,
                  float* Cf, bf16_t* Cb, int M, int K, int N, float scale,
                  int relu) {
    const int waves = (M / 16) * (N / 64);
    gemm_bf16_wmma<<<(waves + 7) / 8, 256, 0, stream>>>(A, Wm, bias, Cf, Cb, M,
                                                        K, N, scale, relu);
  };

  // ---- weight conversion (bf16 staging) ----
  conv(Wc,   wc_bf,   256 * 64);
  conv(Wqkv, wqkv_bf, 4 * 768 * 256);
  conv(Wo,   wo_bf,   4 * 256 * 256);
  conv(W1,   w1_bf,   4 * 128 * 256);
  conv(W2,   w2_bf,   4 * 256 * 128);
  conv(WQ,   wq_bf,   64 * 256);
  conv(WK,   wk_bf,   64 * 256);
  conv(WV,   wv_bf,   64 * 256);

  // ---- embedding: h = LN_F(x) @ Wc^T + bc ----
  ln_feat_kernel<<<TOK / 8, 256, 0, stream>>>(x, g0, b0, hn_bf);
  gemm(hn_bf, wc_bf, bc, h_f32, h_bf, TOK, 64, 256, 1.f, 0);

  // ---- transformer layers (post-norm) ----
  for (int l = 0; l < 4; ++l) {
    gemm(h_bf, wqkv_bf + (size_t)l * 768 * 256, bqkv + l * 768, nullptr,
         qkv_bf, TOK, 256, 768, 1.f, 0);
    attn64_kernel<<<dim3(1024, 8), 128, 0, stream>>>(qkv_bf, o_bf);
    gemm(o_bf, wo_bf + (size_t)l * 256 * 256, bo + l * 256, tmp_f32, nullptr,
         TOK, 256, 256, 1.f, 0);
    ln_res_kernel<<<TOK / 8, 256, 0, stream>>>(h_f32, tmp_f32, ln1g + l * 256,
                                               ln1b + l * 256, h_f32, h_bf);
    gemm(h_bf, w1_bf + (size_t)l * 128 * 256, b1 + l * 128, nullptr, ff1_bf,
         TOK, 256, 128, 1.f, 1);
    gemm(ff1_bf, w2_bf + (size_t)l * 256 * 128, b2 + l * 256, tmp_f32, nullptr,
         TOK, 128, 256, 1.f, 0);
    ln_res_kernel<<<TOK / 8, 256, 0, stream>>>(h_f32, tmp_f32, ln2g + l * 256,
                                               ln2b + l * 256, h_f32, h_bf);
  }

  // ---- mean over sequence -> per-asset representation ----
  mean_s_kernel<<<1024, 256, 0, stream>>>(h_f32, r_bf);

  // ---- CAAN ----
  gemm(r_bf, wq_bf, bQ, nullptr, cq_bf, 1024, 256, 64, 1.f, 0);
  gemm(r_bf, wk_bf, bK, nullptr, ck_bf, 1024, 256, 64, 1.f, 0);
  gemm(r_bf, wv_bf, bV, nullptr, cv_bf, 1024, 256, 64, 1.f, 0);
  // attnS = (q @ k^T) / sqrt(64)
  gemm(cq_bf, ck_bf, nullptr, attnS, nullptr, 1024, 64, 1024, 0.125f, 0);
  softmax1024_kernel<<<1024, 256, 0, stream>>>(attnS, attnP);
  transpose_v_kernel<<<256, 256, 0, stream>>>(cv_bf, cvT_bf);
  gemm(attnP, cvT_bf, nullptr, av, nullptr, 1024, 1024, 64, 1.f, 0);
  score_kernel<<<4, 256, 0, stream>>>(av, Wsw, bsb, scores);

  // ---- top-10 + masked-softmax allocation -> output tuple ----
  topk_alloc_kernel<<<1, 1024, 0, stream>>>(scores, (float*)d_out);
}
